// MSA_mean_10926396801700
// MI455X (gfx1250) — matrix operand order repaired and checked by reading, not compiled, not run on hardware
//
#include <hip/hip_runtime.h>
#include <hip/hip_bf16.h>

// ---------------------------------------------------------------------------
// Nystromformer forward on MI455X (gfx1250, wave32, WMMA).
// All GEMM operands are pre-converted bf16 (weights converted+transposed once,
// activations written bf16 by producer kernels / GEMM epilogue mirrors), so the
// WMMA inner loop is pure data movement + v_wmma_f32_16x16x32_bf16.
// Async global->LDS staging (ASYNCcnt) used when the builtin is available.
// ---------------------------------------------------------------------------

typedef __attribute__((ext_vector_type(16))) __bf16 v16bf;
typedef __attribute__((ext_vector_type(8)))  float  v8f;

union FragBF { uint4 u[2]; v16bf v; };
union AccF   { v8f v; float f[8]; };

__device__ __forceinline__ unsigned short f2bf(float f) {
  unsigned int u = __float_as_uint(f);
  u += 0x7FFFu + ((u >> 16) & 1u);          // round-to-nearest-even
  return (unsigned short)(u >> 16);
}
__device__ __forceinline__ float bf2f(unsigned short u) {
  return __uint_as_float((unsigned int)u << 16);
}

#if defined(__gfx1250__) && __has_builtin(__builtin_amdgcn_global_load_async_to_lds_b128)
#define ASYNC_LDS 1
#else
#define ASYNC_LDS 0
#endif

#if ASYNC_LDS
typedef int vsi4 __attribute__((vector_size(4 * sizeof(int))));
typedef __attribute__((address_space(1))) vsi4* g_v4p;
typedef __attribute__((address_space(3))) vsi4* l_v4p;
__device__ __forceinline__ void async_g2l_b128(const unsigned short* g, unsigned short* l) {
  __builtin_amdgcn_global_load_async_to_lds_b128(
      (g_v4p)(void*)g, (l_v4p)(void*)l, 0, 0);
}
__device__ __forceinline__ void wait_async0() {
#if __has_builtin(__builtin_amdgcn_s_wait_asynccnt)
  __builtin_amdgcn_s_wait_asynccnt(0);
#else
  asm volatile("s_wait_asynccnt 0x0" ::: "memory");
#endif
}
#endif

#define TM 128
#define TN 64
#define TK 32
#define SA 40   // LDS row stride (bf16 elems) for A tile, +8 pad
#define SB 40   // LDS row stride for B tile stored [n][k]

// D = alpha*(A@B or A@B^T) + diag*I + bias + resid ; outputs: fp32 Cf,
// bf16 Cbf (same layout), bf16 CbfT (transposed). Batched over blockIdx.z.
// All M/N/K must be multiples of 128/64/32 (true for every call site).
__global__ __launch_bounds__(256)
void gemm_wmma_bf16_kernel(const unsigned short* __restrict__ A,
                           const unsigned short* __restrict__ B,
                           float* __restrict__ Cf,
                           unsigned short* __restrict__ Cbf,
                           unsigned short* __restrict__ CbfT,
                           const float* __restrict__ bias,
                           const float* __restrict__ resid,
                           int M, int N, int K,
                           int lda, long long sA,
                           int ldb, long long sB,
                           int ldc, long long sC,
                           int ldct, long long sCT,
                           float alpha, float diagVal, int transB, int relu)
{
  __shared__ __align__(16) unsigned short As[TM * SA];
  __shared__ __align__(16) unsigned short Bs[TN * SB];

  const int tid = threadIdx.x;
  const int bz  = blockIdx.z;
  const unsigned short* Ab = A + (long long)bz * sA;
  const unsigned short* Bb = B + (long long)bz * sB;

  const int rowBase = blockIdx.y * TM;
  const int colBase = blockIdx.x * TN;

  const int lane = tid & 31;
  const int wid  = tid >> 5;
  const int wm   = (wid & 3) * 32;
  const int wn   = (wid >> 2) * 32;
  const int l15  = lane & 15;
  const int lhi  = (lane >> 4) & 1;

  v8f acc00 = {}, acc01 = {}, acc10 = {}, acc11 = {};

  const int ar = tid >> 1;             // A stage: 128 rows, 16 elems/thread
  const int ac = (tid & 1) << 4;
  const int nr = tid >> 2;             // B(transB) stage: 64 rows, 8 elems/thread
  const int kc = (tid & 3) << 3;

  for (int k0 = 0; k0 < K; k0 += TK) {
    // ---- stage A tile (TM x TK) bf16, pure copy ----
    {
      const unsigned short* gp = Ab + (long long)(rowBase + ar) * lda + (k0 + ac);
      unsigned short* lp = &As[ar * SA + ac];
#if ASYNC_LDS
      async_g2l_b128(gp, lp);
      async_g2l_b128(gp + 8, lp + 8);
#else
      ((uint4*)lp)[0] = ((const uint4*)gp)[0];
      ((uint4*)lp)[1] = ((const uint4*)gp)[1];
#endif
      if (k0 + TK < K) __builtin_prefetch(gp + TK, 0, 3);   // global_prefetch_b8
    }
    // ---- stage B tile as [n][k] bf16 ----
    if (transB) {                       // B stored N x K: contiguous copy
      const unsigned short* gp = Bb + (long long)(colBase + nr) * ldb + (k0 + kc);
      unsigned short* lp = &Bs[nr * SB + kc];
#if ASYNC_LDS
      async_g2l_b128(gp, lp);
#else
      *(uint4*)lp = *(const uint4*)gp;
#endif
    } else {                            // B row-major K x N: transpose scatter
      int kr = tid >> 3;                // 0..31
      int nc = (tid & 7) << 3;          // 0..56
      const unsigned short* gp = Bb + (long long)(k0 + kr) * ldb + colBase + nc;
      unsigned short vs[8];
      *(uint4*)vs = *(const uint4*)gp;
      #pragma unroll
      for (int j = 0; j < 8; ++j) Bs[(nc + j) * SB + kr] = vs[j];
    }
#if ASYNC_LDS
    wait_async0();
#endif
    __syncthreads();

    // ---- fragments per gfx1250 16-bit WMMA layouts ----
    FragBF a0, a1f, b0, b1f;
    {
      int m0 = wm + l15;
      int kb = lhi * 8;
      a0.u[0]  = *(const uint4*)&As[m0 * SA + kb];
      a0.u[1]  = *(const uint4*)&As[m0 * SA + kb + 16];
      a1f.u[0] = *(const uint4*)&As[(m0 + 16) * SA + kb];
      a1f.u[1] = *(const uint4*)&As[(m0 + 16) * SA + kb + 16];

      int n0  = wn + l15;
      int kbB = lhi * 8;
      b0.u[0]  = *(const uint4*)&Bs[n0 * SB + kbB];
      b0.u[1]  = *(const uint4*)&Bs[n0 * SB + kbB + 16];
      b1f.u[0] = *(const uint4*)&Bs[(n0 + 16) * SB + kbB];
      b1f.u[1] = *(const uint4*)&Bs[(n0 + 16) * SB + kbB + 16];
    }
    acc00 = __builtin_amdgcn_wmma_f32_16x16x32_bf16(false, a0.v,  false, b0.v,  (short)0, acc00, false, false);
    acc01 = __builtin_amdgcn_wmma_f32_16x16x32_bf16(false, a0.v,  false, b1f.v, (short)0, acc01, false, false);
    acc10 = __builtin_amdgcn_wmma_f32_16x16x32_bf16(false, a1f.v, false, b0.v,  (short)0, acc10, false, false);
    acc11 = __builtin_amdgcn_wmma_f32_16x16x32_bf16(false, a1f.v, false, b1f.v, (short)0, acc11, false, false);
    __syncthreads();
  }

  // ---- epilogue ----
  float* Cfb            = Cf   ? Cf   + (long long)bz * sC  : (float*)0;
  unsigned short* Cbfb  = Cbf  ? Cbf  + (long long)bz * sC  : (unsigned short*)0;
  unsigned short* CbfTb = CbfT ? CbfT + (long long)bz * sCT : (unsigned short*)0;
  const float* Rb       = resid ? resid + (long long)bz * sC : (const float*)0;

  AccF av[2][2];
  av[0][0].v = acc00; av[0][1].v = acc01; av[1][0].v = acc10; av[1][1].v = acc11;
  #pragma unroll
  for (int i = 0; i < 2; ++i) {
    #pragma unroll
    for (int j = 0; j < 2; ++j) {
      int gcol = colBase + wn + j * 16 + l15;
      float bval = bias ? bias[gcol] : 0.f;
      #pragma unroll
      for (int r = 0; r < 8; ++r) {
        int grow = rowBase + wm + i * 16 + (lhi ? r + 8 : r);
        float v = av[i][j].f[r] * alpha;
        if (grow == gcol) v += diagVal;
        v += bval;
        long long ci = (long long)grow * ldc + gcol;
        if (Rb) v += Rb[ci];
        if (relu) v = fmaxf(v, 0.f);
        if (Cfb)   Cfb[ci] = v;
        if (Cbfb)  Cbfb[ci] = f2bf(v);
        if (CbfTb) CbfTb[(long long)gcol * ldct + grow] = f2bf(v);
      }
    }
  }
}

// ---------------- fp32 -> bf16 convert (flat) ----------------
__global__ __launch_bounds__(256)
void cvt_bf16_kernel(const float* __restrict__ in, unsigned short* __restrict__ out, long long n)
{
  long long i = (long long)blockIdx.x * 256 + threadIdx.x;
  if (i < n) out[i] = f2bf(in[i]);
}

// fp32 (R x C row-major) -> bf16 transposed (C x R)
__global__ __launch_bounds__(256)
void cvt_bf16_tr_kernel(const float* __restrict__ in, unsigned short* __restrict__ out, int R, int C)
{
  long long i = (long long)blockIdx.x * 256 + threadIdx.x;
  if (i < (long long)R * C) {
    int r = (int)(i / C), c = (int)(i % C);
    out[(long long)c * R + r] = f2bf(in[i]);
  }
}

// ---------------- layernorm: fp32 in -> bf16 out ----------------
__global__ __launch_bounds__(256)
void layernorm_kernel(const float* __restrict__ x, const float* __restrict__ g,
                      const float* __restrict__ b, unsigned short* __restrict__ out, int cols)
{
  long long row = blockIdx.x;
  const float* xr = x + row * cols;
  unsigned short* orow = out + row * cols;
  int t = threadIdx.x;
  __shared__ float red[256];
  float s = 0.f;
  for (int j = t; j < cols; j += 256) s += xr[j];
  red[t] = s; __syncthreads();
  for (int o = 128; o > 0; o >>= 1) { if (t < o) red[t] += red[t + o]; __syncthreads(); }
  float mu = red[0] / cols; __syncthreads();
  float v = 0.f;
  for (int j = t; j < cols; j += 256) { float d = xr[j] - mu; v += d * d; }
  red[t] = v; __syncthreads();
  for (int o = 128; o > 0; o >>= 1) { if (t < o) red[t] += red[t + o]; __syncthreads(); }
  float rstd = rsqrtf(red[0] / cols + 1e-5f);
  for (int j = t; j < cols; j += 256) orow[j] = f2bf((xr[j] - mu) * rstd * g[j] + b[j]);
}

// ---------------- row softmax: fp32 in-place + bf16 mirror ----------------
__global__ __launch_bounds__(256)
void softmax_rows_kernel(float* __restrict__ data, unsigned short* __restrict__ obf, int cols)
{
  long long row = blockIdx.x;
  float* d = data + row * cols;
  unsigned short* ob = obf + row * cols;
  int t = threadIdx.x;
  __shared__ float red[256];
  float m = -3.402823e38f;
  for (int j = t; j < cols; j += 256) m = fmaxf(m, d[j]);
  red[t] = m; __syncthreads();
  for (int o = 128; o > 0; o >>= 1) { if (t < o) red[t] = fmaxf(red[t], red[t + o]); __syncthreads(); }
  m = red[0]; __syncthreads();
  float s = 0.f;
  for (int j = t; j < cols; j += 256) { float e = __expf(d[j] - m); d[j] = e; s += e; }
  red[t] = s; __syncthreads();
  for (int o = 128; o > 0; o >>= 1) { if (t < o) red[t] += red[t + o]; __syncthreads(); }
  float inv = 1.f / red[0];
  for (int j = t; j < cols; j += 256) { float e = d[j] * inv; d[j] = e; ob[j] = f2bf(e); }
}

// ---------------- landmark means (bf16 in, bf16 out) ----------------
__global__ void landmark_kernel(const unsigned short* __restrict__ qkv,
                                unsigned short* __restrict__ ql, unsigned short* __restrict__ kl)
{
  int m = blockIdx.x, h = blockIdx.y, d = threadIdx.x;
  float sq = 0.f, sk = 0.f;
  for (int tt = 0; tt < 64; ++tt) {
    long long r = (long long)(m * 64 + tt) * 1536;
    sq += bf2f(qkv[r + h * 64 + d]);
    sk += bf2f(qkv[r + 512 + h * 64 + d]);
  }
  long long o = ((long long)h * 256 + m) * 64 + d;
  ql[o] = f2bf(sq * (1.f / 64.f));
  kl[o] = f2bf(sk * (1.f / 64.f));
}

// ---------------- pinv: per-head max |row-sum| / |col-sum| ----------------
__global__ __launch_bounds__(256)
void pinv_sums_kernel(const float* __restrict__ a2, float* __restrict__ hrow, float* __restrict__ hcol)
{
  int h = blockIdx.x, t = threadIdx.x;
  const float* base = a2 + (long long)h * 65536;
  float cs = 0.f, rs = 0.f;
  for (int i = 0; i < 256; ++i) {
    cs += fabsf(base[i * 256 + t]);
    rs += fabsf(base[t * 256 + i]);
  }
  __shared__ float rc[256], rr[256];
  rc[t] = cs; rr[t] = rs; __syncthreads();
  for (int o = 128; o > 0; o >>= 1) {
    if (t < o) { rc[t] = fmaxf(rc[t], rc[t + o]); rr[t] = fmaxf(rr[t], rr[t + o]); }
    __syncthreads();
  }
  if (t == 0) { hcol[h] = rc[0]; hrow[h] = rr[0]; }
}

__global__ void pinv_scale_kernel(const float* hrow, const float* hcol, float* scale)
{
  if (threadIdx.x == 0) {
    float a = hrow[0], b = hcol[0];
    for (int h = 1; h < 8; ++h) { a = fmaxf(a, hrow[h]); b = fmaxf(b, hcol[h]); }
    scale[0] = a * b;
  }
}

// Z = a2^T/scale : write bf16 normal (Zbf) and bf16 transposed mirror (ZbfT = a2/scale)
__global__ __launch_bounds__(256)
void pinv_init_kernel(const float* __restrict__ a2, const float* __restrict__ scale,
                      unsigned short* __restrict__ zbf, unsigned short* __restrict__ zbfT)
{
  int i = blockIdx.x, h = blockIdx.y, j = threadIdx.x;
  float inv = 1.f / scale[0];
  long long o = ((long long)h * 256 + i) * 256 + j;
  zbf [o] = f2bf(a2[((long long)h * 256 + j) * 256 + i] * inv);
  zbfT[o] = f2bf(a2[o] * inv);
}

// T1^T = (7I - XZ)^T : t1T[h][i][j] = 7*delta(i,j) - xz[h][j][i]
__global__ __launch_bounds__(256)
void pinv_seven_kernel(const float* __restrict__ xz, unsigned short* __restrict__ t1T)
{
  int i = blockIdx.x & 255, h = blockIdx.x >> 8, j = threadIdx.x;
  float v = (i == j ? 7.f : 0.f) - xz[((long long)h * 256 + j) * 256 + i];
  t1T[((long long)h * 256 + i) * 256 + j] = f2bf(v);
}

// depthwise conv(K=33) residual + conv bias; COMB fp32 in -> COMBbf bf16 out
__global__ __launch_bounds__(256)
void conv_add_kernel(const unsigned short* __restrict__ qkv, const float* __restrict__ cw,
                     const float* __restrict__ cb, const float* __restrict__ comb,
                     unsigned short* __restrict__ combbf, int n)
{
  int h = blockIdx.y;
  int t = blockIdx.x * 256 + threadIdx.x;   // over n*64
  int nn = t >> 6, d = t & 63;
  long long ci = (long long)nn * 512 + h * 64 + d;
  float s = cb[h] + comb[ci];
  #pragma unroll 1
  for (int kk = 0; kk < 33; ++kk) {
    int p = nn + kk - 16;
    if (p >= 0 && p < n) s += bf2f(qkv[(long long)p * 1536 + 1024 + h * 64 + d]) * cw[h * 33 + kk];
  }
  combbf[ci] = f2bf(s);
}

// mean over sequence (grid 512 = cols)
__global__ __launch_bounds__(256)
void meanpool_kernel(const float* __restrict__ x, float* __restrict__ pooled, int n)
{
  int j = blockIdx.x, t = threadIdx.x;
  float s = 0.f;
  for (int i = t; i < n; i += 256) s += x[(long long)i * 512 + j];
  __shared__ float red[256];
  red[t] = s; __syncthreads();
  for (int o = 128; o > 0; o >>= 1) { if (t < o) red[t] += red[t + o]; __syncthreads(); }
  if (t == 0) pooled[j] = red[0] / (float)n;
}

// final head: logits, softmax, argmax -> d_out[0..4]
__global__ void head_kernel(const float* __restrict__ pooled, const float* __restrict__ w,
                            const float* __restrict__ b, float* __restrict__ out)
{
  if (threadIdx.x == 0) {
    float l0 = b[0], l1 = b[1];
    for (int j = 0; j < 512; ++j) { l0 += pooled[j] * w[j * 2 + 0]; l1 += pooled[j] * w[j * 2 + 1]; }
    float m = fmaxf(l0, l1);
    float e0 = __expf(l0 - m), e1 = __expf(l1 - m);
    float s = e0 + e1;
    out[0] = l0; out[1] = l1; out[2] = e0 / s; out[3] = e1 / s;
    out[4] = (l1 > l0) ? 1.f : 0.f;
  }
}

// ---------------------------------------------------------------------------
extern "C" void kernel_launch(void* const* d_in, const int* in_sizes, int n_in,
                              void* d_out, int out_size, void* d_ws, size_t ws_size,
                              hipStream_t stream)
{
  (void)in_sizes; (void)n_in; (void)out_size; (void)ws_size;
  const float* h_in   = (const float*)d_in[0];
  const float* fc1_w  = (const float*)d_in[1];
  const float* fc1_b  = (const float*)d_in[2];
  const float* norm_g = (const float*)d_in[3];
  const float* norm_b = (const float*)d_in[4];
  const float* w_qkv  = (const float*)d_in[5];
  const float* w_out  = (const float*)d_in[6];
  const float* b_out  = (const float*)d_in[7];
  const float* conv_w = (const float*)d_in[8];
  const float* conv_b = (const float*)d_in[9];
  const float* fc2_w  = (const float*)d_in[10];
  const float* fc2_b  = (const float*)d_in[11];

  const int N = 16384, DM = 512, H = 8, M = 256;

  float* wsf = (float*)d_ws;
  long long off = 0;
  auto allocF = [&](long long nf) { float* p = wsf + off; off += nf; return p; };
  auto allocU = [&](long long nu) { unsigned short* p = (unsigned short*)(wsf + off); off += (nu + 1) / 2; return p; };

  float*          X     = allocF((long long)N * DM);
  float*          COMB  = allocF((long long)N * DM);
  float*          SC1   = allocF((long long)H * N * M);     // a1 scores fp32
  float*          SC3   = allocF((long long)H * M * N);     // a3 scores fp32
  float*          A2f   = allocF((long long)H * M * M);
  float*          XZf   = allocF((long long)H * M * M);
  float*          HROW  = allocF(8);
  float*          HCOL  = allocF(8);
  float*          SCL   = allocF(8);
  float*          POOL  = allocF(512);

  unsigned short* Hbf   = allocU((long long)N * 1024);
  unsigned short* W1T   = allocU((long long)DM * 1024);     // fc1_w^T  (512 x 1024)
  unsigned short* WQT   = allocU(2LL * 1536 * DM);          // w_qkv^T per layer (1536 x 512)
  unsigned short* WOT   = allocU(2LL * DM * DM);            // w_out^T per layer
  unsigned short* LNbf  = allocU((long long)N * DM);
  unsigned short* QKVbf = allocU((long long)N * 1536);
  unsigned short* A1bf  = allocU((long long)H * N * M);
  unsigned short* A3bf  = allocU((long long)H * M * N);
  unsigned short* A2bf  = allocU((long long)H * M * M);
  unsigned short* QLbf  = allocU((long long)H * M * 64);
  unsigned short* KLbf  = allocU((long long)H * M * 64);
  unsigned short* Zbf   = allocU((long long)H * M * M);
  unsigned short* ZbfT  = allocU((long long)H * M * M);
  unsigned short* ZNbf  = allocU((long long)H * M * M);
  unsigned short* ZNbfT = allocU((long long)H * M * M);
  unsigned short* XZbf  = allocU((long long)H * M * M);
  unsigned short* T1T   = allocU((long long)H * M * M);
  unsigned short* T2T   = allocU((long long)H * M * M);
  unsigned short* A3VT  = allocU((long long)H * 64 * M);    // (a3@v)^T per head (64 x 256)
  unsigned short* A1Zbf = allocU((long long)H * N * M);
  unsigned short* COMBbf= allocU((long long)N * DM);

  auto gemm = [&](const unsigned short* A, int lda, long long sA,
                  const unsigned short* B, int ldb, long long sB,
                  float* Cf, unsigned short* Cbf, int ldc, long long sC,
                  unsigned short* CbfT, int ldct, long long sCT,
                  const float* bias, const float* resid,
                  int m, int n, int k, int batch,
                  float alpha, float diag, int transB, int relu) {
    dim3 g(n / TN, m / TM, batch);
    gemm_wmma_bf16_kernel<<<g, dim3(256), 0, stream>>>(
        A, B, Cf, Cbf, CbfT, bias, resid, m, n, k,
        lda, sA, ldb, sB, ldc, sC, ldct, sCT, alpha, diag, transB, relu);
  };

  const float qscale = 0.125f;   // DH^-0.5
  const long long sMM = (long long)M * M;

  // ---- one-time input conversions (weights transposed for transB staging) ----
  cvt_bf16_kernel<<<(N * 1024LL + 255) / 256, 256, 0, stream>>>(h_in, Hbf, (long long)N * 1024);
  cvt_bf16_tr_kernel<<<(1024 * 512 + 255) / 256, 256, 0, stream>>>(fc1_w, W1T, 1024, 512);
  for (int l = 0; l < 2; ++l) {
    cvt_bf16_tr_kernel<<<(512 * 1536 + 255) / 256, 256, 0, stream>>>(
        w_qkv + (long long)l * 512 * 1536, WQT + (long long)l * 1536 * 512, 512, 1536);
    cvt_bf16_tr_kernel<<<(512 * 512 + 255) / 256, 256, 0, stream>>>(
        w_out + (long long)l * 512 * 512, WOT + (long long)l * 512 * 512, 512, 512);
  }

  // fc1 + relu : X = relu(h @ fc1_w + fc1_b)
  gemm(Hbf, 1024, 0, W1T, 1024, 0, X, nullptr, DM, 0, nullptr, 0, 0,
       fc1_b, nullptr, N, DM, 1024, 1, 1.f, 0.f, 1, 1);

  for (int layer = 0; layer < 2; ++layer) {
    layernorm_kernel<<<N, 256, 0, stream>>>(X, norm_g + layer * DM, norm_b + layer * DM, LNbf, DM);
    // QKV (bf16 only)
    gemm(LNbf, DM, 0, WQT + (long long)layer * 1536 * DM, DM, 0,
         nullptr, QKVbf, 1536, 0, nullptr, 0, 0, nullptr, nullptr,
         N, 1536, DM, 1, 1.f, 0.f, 1, 0);
    landmark_kernel<<<dim3(M, H), 64, 0, stream>>>(QKVbf, QLbf, KLbf);

    // a1 = softmax( q*scale @ kl^T )  [H,N,M]
    gemm(QKVbf, 1536, 64, KLbf, 64, (long long)M * 64, SC1, nullptr, M, (long long)N * M,
         nullptr, 0, 0, nullptr, nullptr, N, M, 64, H, qscale, 0.f, 1, 0);
    softmax_rows_kernel<<<H * N, 256, 0, stream>>>(SC1, A1bf, M);

    // a2 = softmax( ql*scale @ kl^T ) [H,M,M] (fp32 + bf16)
    gemm(QLbf, 64, (long long)M * 64, KLbf, 64, (long long)M * 64, A2f, nullptr, M, sMM,
         nullptr, 0, 0, nullptr, nullptr, M, M, 64, H, qscale, 0.f, 1, 0);
    softmax_rows_kernel<<<H * M, 256, 0, stream>>>(A2f, A2bf, M);

    // a3 = softmax( ql*scale @ k^T )  [H,M,N]
    gemm(QLbf, 64, (long long)M * 64, QKVbf + 512, 1536, 64, SC3, nullptr, N, (long long)M * N,
         nullptr, 0, 0, nullptr, nullptr, M, N, 64, H, qscale, 0.f, 1, 0);
    softmax_rows_kernel<<<H * M, 256, 0, stream>>>(SC3, A3bf, N);

    // ---- Moore-Penrose pinv (Newton-Schulz, bf16 WMMA) ----
    pinv_sums_kernel<<<H, 256, 0, stream>>>(A2f, HROW, HCOL);
    pinv_scale_kernel<<<1, 32, 0, stream>>>(HROW, HCOL, SCL);
    pinv_init_kernel<<<dim3(M, H), 256, 0, stream>>>(A2f, SCL, Zbf, ZbfT);
    unsigned short *Zc = Zbf, *ZcT = ZbfT, *Zn = ZNbf, *ZnT = ZNbfT;
    for (int it = 0; it < 6; ++it) {
      // XZ = a2 @ Z  (fp32 for elementwise + bf16 as next A)
      gemm(A2bf, M, sMM, ZcT, M, sMM, XZf, XZbf, M, sMM, nullptr, 0, 0,
           nullptr, nullptr, M, M, M, H, 1.f, 0.f, 1, 0);
      pinv_seven_kernel<<<H * M, 256, 0, stream>>>(XZf, T1T);          // T1 = 7I - XZ (transposed)
      // T2 = 15I - XZ@T1  -> transposed bf16 only
      gemm(XZbf, M, sMM, T1T, M, sMM, nullptr, nullptr, M, sMM, T2T, M, sMM,
           nullptr, nullptr, M, M, M, H, -1.f, 15.f, 1, 0);
      // T1 = 13I - XZ@T2  -> transposed bf16 only
      gemm(XZbf, M, sMM, T2T, M, sMM, nullptr, nullptr, M, sMM, T1T, M, sMM,
           nullptr, nullptr, M, M, M, H, -1.f, 13.f, 1, 0);
      // Zn = 0.25 * Z@T1  -> normal + transposed bf16
      gemm(Zc, M, sMM, T1T, M, sMM, nullptr, Zn, M, sMM, ZnT, M, sMM,
           nullptr, nullptr, M, M, M, H, 0.25f, 0.f, 1, 0);
      unsigned short* t;
      t = Zc; Zc = Zn; Zn = t;
      t = ZcT; ZcT = ZnT; ZnT = t;
    }

    // a3v = a3 @ v  -> transposed bf16 (64 x 256 per head)
    gemm(A3bf, N, (long long)M * N, QKVbf + 1024, 1536, 64,
         nullptr, nullptr, 64, (long long)M * 64, A3VT, M, (long long)64 * M,
         nullptr, nullptr, M, 64, N, H, 1.f, 0.f, 0, 0);
    // a1z = a1 @ pinv(a2) -> bf16 [H,N,M]
    gemm(A1bf, M, (long long)N * M, ZcT, M, sMM,
         nullptr, A1Zbf, M, (long long)N * M, nullptr, 0, 0,
         nullptr, nullptr, N, M, M, H, 1.f, 0.f, 1, 0);
    // COMB[n, h*64+d] = a1z @ a3v (fp32)
    gemm(A1Zbf, M, (long long)N * M, A3VT, M, (long long)64 * M,
         COMB, nullptr, DM, 64, nullptr, 0, 0,
         nullptr, nullptr, N, 64, M, H, 1.f, 0.f, 1, 0);
    // += depthwise conv residual + bias -> bf16
    conv_add_kernel<<<dim3(N * 64 / 256, H), 256, 0, stream>>>(
        QKVbf, conv_w + (long long)layer * H * 33, conv_b + (long long)layer * H, COMB, COMBbf, N);
    // X = X + COMBbf @ w_out + b_out (fused residual, fp32 out)
    gemm(COMBbf, DM, 0, WOT + (long long)layer * DM * DM, DM, 0,
         X, nullptr, DM, 0, nullptr, 0, 0,
         b_out + (long long)layer * DM, X, N, DM, DM, 1, 1.f, 0.f, 1, 0);
  }

  meanpool_kernel<<<DM, 256, 0, stream>>>(X, POOL, N);
  head_kernel<<<1, 32, 0, stream>>>(POOL, fc2_w, fc2_b, (float*)d_out);
}
